// DatasetProjector_13383118094957
// MI455X (gfx1250) — matrix operand order repaired and checked by reading, not compiled, not run on hardware
//
#include <hip/hip_runtime.h>
#include <hip/hip_bf16.h>
#include <cmath>

// ---------------------------------------------------------------------------
// DatasetProjector fused kernel for gfx1250 (MI455X):
//   pad(ch) -> Conv1d(256->4096,k=3,p=1) -> +bias -> GELU(erf) -> select slice
// fused as: per-batch implicit GEMM over ONLY the selected 512 channels,
// contraction on v_wmma_f32_16x16x32_bf16 with fp32 accumulation.
// ---------------------------------------------------------------------------

typedef __attribute__((ext_vector_type(8)))  __bf16 v8bf;
typedef __attribute__((ext_vector_type(16))) __bf16 v16bf;
typedef __attribute__((ext_vector_type(8)))  float  v8f;

#define SENSOR_REAL 200     // real input channels (200..255 are zero padding)
#define C_USED      224     // contracted channels, rounded to multiple of 32
#define CROW        232     // LDS row stride in bf16 (224 + 8 pad -> bank-safe)
#define T_TILE      128
#define M_TILE      64
#define XS_ROWS     (T_TILE + 2)              // halo for k=3, pad=1
#define XS_ELEMS    (XS_ROWS * CROW)          // 30160 bf16
#define WS_PLANE    (M_TILE * CROW)           // 14848 bf16 per tap
#define WS_ELEMS    (3 * WS_PLANE)            // 44544 bf16
#define SMEM_ELEMS  (XS_ELEMS + WS_ELEMS)
#define SMEM_BYTES  (SMEM_ELEMS * 2)          // 149408 B  (<=320KB WGP LDS)

static __device__ __forceinline__ unsigned short f2bf(float f) {
  unsigned u = __float_as_uint(f);
  u += 0x7FFFu + ((u >> 16) & 1u);            // round-to-nearest-even
  return (unsigned short)(u >> 16);
}

static __device__ __forceinline__ v16bf ld_frag(const unsigned short* p0,
                                                const unsigned short* p1) {
  union { v16bf v; v8bf h[2]; } u;
  u.h[0] = *(const v8bf*)p0;                  // ds_load_b128
  u.h[1] = *(const v8bf*)p1;                  // ds_load_b128
  return u.v;
}

static __device__ __forceinline__ float gelu_exact(float x) {
  return 0.5f * x * (1.0f + erff(x * 0.70710678118654752f));
}

static __device__ __forceinline__ void store_tile(float* out_b, int m_g,
                                                  int tbase, float biasv,
                                                  const v8f& acc) {
  float* p = out_b + (size_t)m_g * 1024 + tbase;
  float4 lo, hi;
  lo.x = gelu_exact(acc[0] + biasv);
  lo.y = gelu_exact(acc[1] + biasv);
  lo.z = gelu_exact(acc[2] + biasv);
  lo.w = gelu_exact(acc[3] + biasv);
  hi.x = gelu_exact(acc[4] + biasv);
  hi.y = gelu_exact(acc[5] + biasv);
  hi.z = gelu_exact(acc[6] + biasv);
  hi.w = gelu_exact(acc[7] + biasv);
  *(float4*)(p)     = lo;                     // global_store_b128
  *(float4*)(p + 4) = hi;                     // global_store_b128
}

__global__ __launch_bounds__(256) void
dsproj_conv_wmma(const float* __restrict__ x,        // [32,200,1024]
                 const int*   __restrict__ dataset_id,
                 const float* __restrict__ W,        // [4096,256,3]
                 const float* __restrict__ bias,     // [4096]
                 float*       __restrict__ out) {    // [32,512,1024]
  extern __shared__ unsigned short smem[];
  unsigned short* xs = smem;                 // xs[tt][c], tt maps t0-1..t0+128
  unsigned short* ws = smem + XS_ELEMS;      // ws[k][m][c]

  const int t0     = blockIdx.x * T_TILE;
  const int mblock = blockIdx.y * M_TILE;
  const int b      = blockIdx.z;
  const int d      = dataset_id[b];
  const int tid    = threadIdx.x;

  // ---- stage x tile (fp32 -> bf16); zero channel pad + time halo ----
  const float* xb = x + (size_t)b * SENSOR_REAL * 1024;
  for (int idx = tid; idx < C_USED * XS_ROWS; idx += 256) {
    int c  = idx / XS_ROWS;
    int tt = idx - c * XS_ROWS;              // lanes walk contiguous t
    int gt = t0 - 1 + tt;
    float v = 0.0f;
    if (c < SENSOR_REAL && gt >= 0 && gt < 1024) v = xb[c * 1024 + gt];
    xs[tt * CROW + c] = f2bf(v);
  }

  // ---- stage selected W slice: ws[k][m][c] (taps de-interleaved) ----
  const float* Wd = W + (size_t)(d * 512 + mblock) * 768;
  for (int idx = tid; idx < M_TILE * (C_USED * 3); idx += 256) {
    int m = idx / (C_USED * 3);
    int q = idx - m * (C_USED * 3);          // q = c*3 + k, contiguous reads
    int c = q / 3;
    int k = q - c * 3;
    ws[k * WS_PLANE + m * CROW + c] = f2bf(Wd[m * 768 + q]);
  }
  __syncthreads();

  const int lane = tid & 31;
  const int wave = tid >> 5;                 // 8 waves (wave32)
  const int tw   = wave & 3;                 // 32-row t strip
  const int mw   = wave >> 2;                // 32-col m half
  const int lrow = lane & 15;
  const int hi   = lane >> 4;

  v8f acc00 = {}, acc01 = {}, acc10 = {}, acc11 = {};

  for (int k = 0; k < 3; ++k) {
    const unsigned short* wp = ws + k * WS_PLANE;
    const int ttA0 = (tw * 32 +  0 + lrow + k) * CROW;   // shift by tap
    const int ttA1 = (tw * 32 + 16 + lrow + k) * CROW;
    const int mB0  = (mw * 32 +  0 + lrow) * CROW;
    const int mB1  = (mw * 32 + 16 + lrow) * CROW;
#pragma unroll
    for (int c0 = 0; c0 < C_USED; c0 += 32) {
      // A frags: halves 0-7 -> K=c0+hi*8+0..7, halves 8-15 -> K=c0+16+hi*8+0..7
      const unsigned short* a0p = xs + ttA0 + c0 + hi * 8;
      const unsigned short* a1p = xs + ttA1 + c0 + hi * 8;
      v16bf a0 = ld_frag(a0p, a0p + 16);
      v16bf a1 = ld_frag(a1p, a1p + 16);
      // B frags: halves 0-7 -> K=c0+hi*8+0..7, halves 8-15 -> K=c0+16+hi*8+0..7
      const unsigned short* b0p = wp + mB0 + c0 + hi * 8;
      const unsigned short* b1p = wp + mB1 + c0 + hi * 8;
      v16bf b0 = ld_frag(b0p, b0p + 16);
      v16bf b1 = ld_frag(b1p, b1p + 16);

      acc00 = __builtin_amdgcn_wmma_f32_16x16x32_bf16(false, a0, false, b0,
                                                      (short)0, acc00, false, false);
      acc01 = __builtin_amdgcn_wmma_f32_16x16x32_bf16(false, a0, false, b1,
                                                      (short)0, acc01, false, false);
      acc10 = __builtin_amdgcn_wmma_f32_16x16x32_bf16(false, a1, false, b0,
                                                      (short)0, acc10, false, false);
      acc11 = __builtin_amdgcn_wmma_f32_16x16x32_bf16(false, a1, false, b1,
                                                      (short)0, acc11, false, false);
    }
  }

  // ---- epilogue: bias + exact GELU; 8 contiguous t per lane -> b128 stores
  float* out_b = out + (size_t)b * 512 * 1024;
  const int m0 = mblock + mw * 32 + lrow;
  const int m1 = m0 + 16;
  const float bias0 = bias[(size_t)d * 512 + m0];
  const float bias1 = bias[(size_t)d * 512 + m1];
  const int tb0 = t0 + tw * 32 + hi * 8;
  const int tb1 = tb0 + 16;

  store_tile(out_b, m0, tb0, bias0, acc00);
  store_tile(out_b, m1, tb0, bias1, acc01);
  store_tile(out_b, m0, tb1, bias0, acc10);
  store_tile(out_b, m1, tb1, bias1, acc11);
}

extern "C" void kernel_launch(void* const* d_in, const int* in_sizes, int n_in,
                              void* d_out, int out_size, void* d_ws, size_t ws_size,
                              hipStream_t stream) {
  const float* x    = (const float*)d_in[0];   // [32,200,1024] f32
  const int*   did  = (const int*)d_in[1];     // [32] i32
  const float* W    = (const float*)d_in[2];   // [4096,256,3] f32
  const float* bv   = (const float*)d_in[3];   // [4096] f32
  float*       out  = (float*)d_out;           // [32,512,1024] f32

  // allow >64KB dynamic LDS (gfx1250 WGP has 320KB); not a stream op
  (void)hipFuncSetAttribute((const void*)dsproj_conv_wmma,
                            hipFuncAttributeMaxDynamicSharedMemorySize,
                            SMEM_BYTES);

  dim3 grid(1024 / T_TILE, 512 / M_TILE, 32);  // (8, 8, 32)
  dsproj_conv_wmma<<<grid, dim3(256), SMEM_BYTES, stream>>>(x, did, W, bv, out);
}
